// prefrontal_model_74955769250574
// MI455X (gfx1250) — matrix operand order repaired and checked by reading, not compiled, not run on hardware
//
#include <hip/hip_runtime.h>

// ---- problem constants (fixed by the reference's setup_inputs) ----
#define T_WIN   19
#define NSTEPS  (T_WIN*T_WIN)   // 361
#define IN1     240
#define IN2     18
#define IN3     42
#define H1      1024
#define H3      8192
#define O1      6
#define O3      16
#define THRESH  0.5f
#define DECAY   0.2f

// ---- workspace layout (float offsets into state region, then byte offsets) ----
#define OFF_H11M  0
#define OFF_H11S  1024
#define OFF_H21M  2048
#define OFF_H21S  3072
#define OFF_H31M  4096
#define OFF_H31S  12288
#define OFF_H315M 20480
#define OFF_H315S 28672
#define OFF_H12M  36864
#define OFF_H12S  36872
#define OFF_H22M  36880
#define OFF_H22S  36888
#define OFF_H32M  36896
#define OFF_H32S  36912
#define OFF_S1    36928
#define OFF_S2    36936
#define OFF_S3    36944
#define OFF_L3    36960
#define STATE_BYTES      262144          // zero this much each call
#define SPIKEBF_BYTE_OFF 163840          // 8192 bf16 spike vector for WMMA A
#define WBF_BYTE_OFF     262144          // swizzled bf16 W315 tiles (128 MB)

typedef __attribute__((ext_vector_type(16))) __bf16 v16bf;
typedef __attribute__((ext_vector_type(8)))  __bf16 v8bf;
typedef __attribute__((ext_vector_type(8)))  float  v8f;

__device__ __forceinline__ float wred(float v) {
  #pragma unroll
  for (int o = 16; o; o >>= 1) v += __shfl_down(v, o, 32);
  return v;
}

// Zero the state region (262144 bytes = 65536 dwords)
__global__ void k_init(unsigned* S) {
  unsigned g = blockIdx.x * blockDim.x + threadIdx.x;
  S[g] = 0u;
}

// One-time: convert W315 fp32 -> bf16 and swizzle into per-wave WMMA B tiles.
// Tile (nt, kc) covers rows nt*16..+15, cols kc*32..+31. Within a tile, lane L
// holds 16 contiguous bf16: row nt*16 + (L&15), cols kc*32 + (L>>4)*16 .. +15
// (contiguous k order per lane half; k3_wmma builds A the same way), stored lane-major so
// a wave's tile read is one contiguous, fully coalesced 1 KB block.
__global__ void k_swizzle(const float* __restrict__ W, unsigned short* __restrict__ Wb) {
  unsigned g    = blockIdx.x * blockDim.x + threadIdx.x;   // 512*256*32 threads
  unsigned lane = g & 31;
  unsigned tile = g >> 5;
  unsigned kc   = tile & 255;
  unsigned nt   = tile >> 8;
  unsigned row  = nt * 16 + (lane & 15);
  unsigned col  = kc * 32 + (lane >> 4) * 16;
  const float* src = W + (size_t)row * H3 + col;
  __bf16* dst = (__bf16*)(Wb + (size_t)tile * 512 + lane * 16);
  #pragma unroll
  for (int e = 0; e < 16; ++e) dst[e] = (__bf16)src[e];
}

// K1: blocks 0..7 compute h11 (1024x240) and h21 (1024x18) for the current t.
//     block 8 finishes the PREVIOUS step: h32 (16x8192, warp-reduced) + s1/s2/s3.
__global__ void k1_front(const float* __restrict__ in2, const float* __restrict__ in3,
                         const float* __restrict__ W11, const float* __restrict__ b11,
                         const float* __restrict__ W21, const float* __restrict__ b21,
                         const float* __restrict__ W32, const float* __restrict__ b32,
                         float* __restrict__ S, int t_cur, int t_prev,
                         int do_cur, int do_fin, int write_out, float* __restrict__ out) {
  int blk = blockIdx.x, tid = threadIdx.x;
  if (blk < 8) {
    if (!do_cur) return;
    int r = blk * 256 + tid;
    if (r < H1) {
      const float* w = W11 + (size_t)r * IN1;
      float acc = 0.f;
      for (int i = 0; i < IN1; ++i) acc += w[i] * in2[i * T_WIN + t_cur];
      float m = S[OFF_H11M + r] * DECAY * (1.f - S[OFF_H11S + r]) + acc + b11[r];
      S[OFF_H11M + r] = m;
      S[OFF_H11S + r] = (m > THRESH) ? 1.f : 0.f;
    } else {
      int q = r - H1;
      const float* w = W21 + (size_t)q * IN2;
      float acc = 0.f;
      #pragma unroll
      for (int i = 0; i < IN2; ++i) acc += w[i] * in3[i * T_WIN + t_cur];
      float m = S[OFF_H21M + q] * DECAY * (1.f - S[OFF_H21S + q]) + acc + b21[q];
      S[OFF_H21M + q] = m;
      S[OFF_H21S + q] = (m > THRESH) ? 1.f : 0.f;
    }
  } else {
    if (!do_fin) return;
    int lane = tid & 31, wrp = tid >> 5;          // 8 warps, 2 h32 rows each
    float keep = (t_prev == 0) ? 0.f : 1.f;       // sums reset when ts[t]==0
    #pragma unroll
    for (int rr = 0; rr < 2; ++rr) {
      int r = wrp + rr * 8;
      const float* wrow = W32 + (size_t)r * H3;
      float p = 0.f;
      for (int i = lane; i < H3; i += 32) p += wrow[i] * S[OFF_H315S + i];
      p = wred(p);
      if (lane == 0) {
        float m = S[OFF_H32M + r] * DECAY * (1.f - S[OFF_H32S + r]) + p + b32[r];
        float s = (m > THRESH) ? 1.f : 0.f;
        S[OFF_H32M + r] = m; S[OFF_H32S + r] = s;
        float sv = S[OFF_S3 + r] * keep + s;
        S[OFF_S3 + r] = sv;
        if (write_out) out[12 + r] = sv;
      }
    }
    if (tid < O1) {
      float sv = S[OFF_S1 + tid] * keep + S[OFF_H12S + tid];
      S[OFF_S1 + tid] = sv;
      if (write_out) out[tid] = sv;
    } else if (tid < 2 * O1) {
      int i = tid - O1;
      float sv = S[OFF_S2 + i] * keep + S[OFF_H22S + i];
      S[OFF_S2 + i] = sv;
      if (write_out) out[6 + i] = sv;
    }
  }
}

// K2a (1 block, 384 threads): h12 = W12@h11s, h22 = W22@h21s (warp per row),
// then build l3 = [x11, x11, h12s, h22s, in3[12:18], in4] (42 floats).
__global__ void k2a(const float* __restrict__ in11, const float* __restrict__ in3,
                    const float* __restrict__ in4,
                    const float* __restrict__ W12, const float* __restrict__ b12,
                    const float* __restrict__ W22, const float* __restrict__ b22,
                    float* __restrict__ S, int t) {
  int tid = threadIdx.x, lane = tid & 31, wrp = tid >> 5;
  if (wrp < 12) {
    int r = (wrp < 6) ? wrp : wrp - 6;
    const float* wrow = ((wrp < 6) ? W12 : W22) + (size_t)r * H1;
    const float* x = S + ((wrp < 6) ? OFF_H11S : OFF_H21S);
    float p = 0.f;
    for (int i = lane; i < H1; i += 32) p += wrow[i] * x[i];
    p = wred(p);
    if (lane == 0) {
      int mo = ((wrp < 6) ? OFF_H12M : OFF_H22M) + r;
      int so = ((wrp < 6) ? OFF_H12S : OFF_H22S) + r;
      float bb = (wrp < 6) ? b12[r] : b22[r];
      float m = S[mo] * DECAY * (1.f - S[so]) + p + bb;
      S[mo] = m; S[so] = (m > THRESH) ? 1.f : 0.f;
    }
  }
  __syncthreads();
  if (tid < IN3) {
    float v;
    if      (tid < 6)  v = in11[tid * T_WIN + t];
    else if (tid < 12) v = in11[(tid - 6) * T_WIN + t];
    else if (tid < 18) v = S[OFF_H12S + tid - 12];
    else if (tid < 24) v = S[OFF_H22S + tid - 18];
    else if (tid < 30) v = in3[(12 + tid - 24) * T_WIN + t];
    else               v = in4[(tid - 30) * T_WIN + t];
    S[OFF_L3 + tid] = v;
  }
}

// K2b (8192 threads): h31 = W31 @ l3; writes fp32 state + bf16 spike vector.
__global__ void k2b(const float* __restrict__ W31, const float* __restrict__ b31,
                    float* __restrict__ S, unsigned short* __restrict__ spk) {
  int r = blockIdx.x * blockDim.x + threadIdx.x;
  const float* w = W31 + (size_t)r * IN3;
  float acc = 0.f;
  #pragma unroll
  for (int i = 0; i < IN3; ++i) acc += w[i] * S[OFF_L3 + i];
  float m = S[OFF_H31M + r] * DECAY * (1.f - S[OFF_H31S + r]) + acc + b31[r];
  float s = (m > THRESH) ? 1.f : 0.f;
  S[OFF_H31M + r] = m; S[OFF_H31S + r] = s;
  ((__bf16*)spk)[r] = (__bf16)s;
}

// K3 (2048 waves, 256 WGs x 256 thr): h315 = W315 @ h31s via
// v_wmma_f32_16x16x32_bf16. K dimension split 4 ways across waves for 4x the
// memory-level parallelism (this kernel is pure L2 bandwidth); partial rows are
// combined through LDS within the workgroup. A = 32-spike chunk replicated
// across all 16 A-rows (all D rows equal; row 0 = GEMV result).
__global__ void k3_wmma(const unsigned short* __restrict__ Wb,
                        const unsigned short* __restrict__ spk,
                        const float* __restrict__ b315, float* __restrict__ S) {
  __shared__ float red[8][16];
  int tid  = threadIdx.x;
  int lane = tid & 31;
  int wv   = tid >> 5;                 // 0..7
  int tloc = wv >> 2;                  // 0..1: n-tile within WG
  int kq   = wv & 3;                   // k-quarter
  int nt   = blockIdx.x * 2 + tloc;    // 0..511
  int half = lane >> 4;
  v8f c = {};
  const __bf16* bp = (const __bf16*)Wb + ((size_t)nt * 256 + (size_t)kq * 64) * 512
                                       + (size_t)lane * 16;
  const __bf16* sp = (const __bf16*)spk + kq * 64 * 32;
  #pragma unroll 4
  for (int kc = 0; kc < 64; ++kc) {
    // A layout (16-bit A 16x32): lane half selects a contiguous K run of 16 (matches B tiles).
    const __bf16* sa = sp + kc * 32 + half * 16;
    v8bf lo = *(const v8bf*)sa;
    v8bf hi = *(const v8bf*)(sa + 8);
    v16bf a = __builtin_shufflevector(lo, hi, 0,1,2,3,4,5,6,7,8,9,10,11,12,13,14,15);
    v16bf b = *(const v16bf*)bp;
    bp += 512;
    c = __builtin_amdgcn_wmma_f32_16x16x32_bf16(false, a, false, b, (short)0, c, false, false);
  }
  if (lane < 16) red[wv][lane] = c[0];
  __syncthreads();
  if (kq == 0 && lane < 16) {
    int j = nt * 16 + lane;
    float dot = red[wv][lane] + red[wv + 1][lane] + red[wv + 2][lane] + red[wv + 3][lane];
    float m = S[OFF_H315M + j] * DECAY * (1.f - S[OFF_H315S + j]) + dot + b315[j];
    S[OFF_H315M + j] = m;
    S[OFF_H315S + j] = (m > THRESH) ? 1.f : 0.f;
  }
}

// Fallback (ws too small for bf16 tiles): fp32 warp-per-row GEMV.
__global__ void k3_f32(const float* __restrict__ W315, const float* __restrict__ b315,
                       float* __restrict__ S) {
  int g = blockIdx.x * blockDim.x + threadIdx.x;
  int lane = g & 31, row = g >> 5;                // 8192 rows
  const float* w = W315 + (size_t)row * H3;
  float p = 0.f;
  for (int i = lane; i < H3; i += 32) p += w[i] * S[OFF_H31S + i];
  p = wred(p);
  if (lane == 0) {
    float m = S[OFF_H315M + row] * DECAY * (1.f - S[OFF_H315S + row]) + p + b315[row];
    S[OFF_H315M + row] = m;
    S[OFF_H315S + row] = (m > THRESH) ? 1.f : 0.f;
  }
}

extern "C" void kernel_launch(void* const* d_in, const int* in_sizes, int n_in,
                              void* d_out, int out_size, void* d_ws, size_t ws_size,
                              hipStream_t stream) {
  const float* in11 = (const float*)d_in[0];
  const float* in2  = (const float*)d_in[2];
  const float* in3  = (const float*)d_in[3];
  const float* in4  = (const float*)d_in[4];
  const float* W11  = (const float*)d_in[5];  const float* b11 = (const float*)d_in[6];
  const float* W12  = (const float*)d_in[7];  const float* b12 = (const float*)d_in[8];
  const float* W21  = (const float*)d_in[9];  const float* b21 = (const float*)d_in[10];
  const float* W22  = (const float*)d_in[11]; const float* b22 = (const float*)d_in[12];
  const float* W31  = (const float*)d_in[13]; const float* b31 = (const float*)d_in[14];
  const float* W315 = (const float*)d_in[15]; const float* b315= (const float*)d_in[16];
  const float* W32  = (const float*)d_in[17]; const float* b32 = (const float*)d_in[18];

  float* S = (float*)d_ws;
  unsigned short* spk = (unsigned short*)((char*)d_ws + SPIKEBF_BYTE_OFF);
  unsigned short* Wbf = (unsigned short*)((char*)d_ws + WBF_BYTE_OFF);
  float* out = (float*)d_out;
  bool use_bf = ws_size >= (size_t)WBF_BYTE_OFF + (size_t)H3 * H3 * 2;

  k_init<<<256, 256, 0, stream>>>((unsigned*)S);
  if (use_bf) k_swizzle<<<16384, 256, 0, stream>>>(W315, Wbf);

  for (int tau = 0; tau < NSTEPS; ++tau) {
    int t  = tau % T_WIN;
    int tp = (tau > 0) ? ((tau - 1) % T_WIN) : 0;
    k1_front<<<9, 256, 0, stream>>>(in2, in3, W11, b11, W21, b21, W32, b32,
                                    S, t, tp, 1, (tau > 0) ? 1 : 0, 0, out);
    k2a<<<1, 384, 0, stream>>>(in11, in3, in4, W12, b12, W22, b22, S, t);
    k2b<<<32, 256, 0, stream>>>(W31, b31, S, spk);
    if (use_bf) k3_wmma<<<256, 256, 0, stream>>>(Wbf, spk, b315, S);
    else        k3_f32<<<1024, 256, 0, stream>>>(W315, b315, S);
  }
  // finish the last step (h32 + sums) and write the 28 outputs
  k1_front<<<9, 256, 0, stream>>>(in2, in3, W11, b11, W21, b21, W32, b32,
                                  S, 0, (NSTEPS - 1) % T_WIN, 0, 1, 1, out);
}
